// Social_Aggregator_70514773066431
// MI455X (gfx1250) — matrix operand order repaired and checked by reading, not compiled, not run on hardware
//
#include <hip/hip_runtime.h>

// ---------------------------------------------------------------------------
// Types for CDNA5 WMMA (wave32): v_wmma_f32_16x16x32_bf16
// ---------------------------------------------------------------------------
typedef __bf16 v16bf __attribute__((ext_vector_type(16)));
typedef __bf16 v8bf  __attribute__((ext_vector_type(8)));
typedef float  v8f   __attribute__((ext_vector_type(8)));

__device__ __forceinline__ unsigned short f32_to_bf16_rne(float f) {
  unsigned int u = __float_as_uint(f);
  u += 0x7FFFu + ((u >> 16) & 1u);          // round-to-nearest-even
  return (unsigned short)(u >> 16);
}

__device__ __forceinline__ v16bf cat8(v8bf lo, v8bf hi) {
  return __builtin_shufflevector(lo, hi, 0, 1, 2, 3, 4, 5, 6, 7,
                                         8, 9, 10, 11, 12, 13, 14, 15);
}

// ---------------------------------------------------------------------------
// Prep: convert W1 [256x128] and W2 [128x128] f32 -> bf16, pre-swizzled into
// per-lane WMMA B-fragment order:
//   frag index = ((kchunk*8 + ntile)*32 + lane)*16 + e
//   value      = W[kchunk*32 + (lane<16 ? 0 : 16) + e][ntile*16 + (lane&15)]
// B-matrix 32x16 bf16, contiguous K order used here (same as A): lanes 0-15 hold K=0..15 for col N,
// lanes 16-31 hold K=16..31.  W1 frags: 32768 bf16 (64KB). W2: 16384 (32KB).
// ---------------------------------------------------------------------------
__global__ void prep_weights(const float* __restrict__ W1,
                             const float* __restrict__ W2,
                             unsigned short* __restrict__ w1f,
                             unsigned short* __restrict__ w2f) {
  const int idx = blockIdx.x * blockDim.x + threadIdx.x;
  if (idx < 32768) {
    const int e    = idx & 15;
    const int lane = (idx >> 4) & 31;
    const int nt   = (idx >> 9) & 7;
    const int kc   = idx >> 12;                    // 0..7
    const int K    = kc * 32 + ((lane < 16) ? 0 : 16) + e;
    const int col  = nt * 16 + (lane & 15);
    w1f[idx] = f32_to_bf16_rne(W1[K * 128 + col]);
  } else if (idx < 32768 + 16384) {
    const int j    = idx - 32768;
    const int e    = j & 15;
    const int lane = (j >> 4) & 31;
    const int nt   = (j >> 9) & 7;
    const int kc   = j >> 12;                      // 0..3
    const int K    = kc * 32 + ((lane < 16) ? 0 : 16) + e;
    const int col  = nt * 16 + (lane & 15);
    w2f[j] = f32_to_bf16_rne(W2[K * 128 + col]);
  }
}

// ---------------------------------------------------------------------------
// Main kernel: one block per node, 256 threads = 8 wave32s.
// Wave w owns output N-tile w (16 cols); two M-tiles cover the 32 neighbors.
// ---------------------------------------------------------------------------
#define XP 136   // ushort pitch for bf16 LDS tiles (272B row -> 4-bank shift)
#define FP 132   // float pitch for f32 LDS tiles  (528B row -> 4-bank shift)

__global__ __launch_bounds__(256)
void social_agg_wmma(const float* __restrict__ emb,
                     const int* __restrict__ nodes,
                     const int* __restrict__ nidx,
                     const unsigned short* __restrict__ w1f,
                     const float* __restrict__ b1,
                     const unsigned short* __restrict__ w2f,
                     const float* __restrict__ b2,
                     const float* __restrict__ W3,
                     const float* __restrict__ b3,
                     float* __restrict__ out) {
  __shared__ float          s_neigh[32][FP];   // gathered neighbor rows (f32)
  __shared__ unsigned short s_xbf[32][XP];     // neighbor rows as bf16
  __shared__ unsigned short s_nodebf[128];     // node embedding as bf16
  __shared__ unsigned short s_hbf[32][XP];     // layer-1 activations bf16
  __shared__ float          s_h2[32][FP];      // layer-2 activations f32
  __shared__ float          s_att[32];         // softmax weights

  const int  n      = blockIdx.x;
  const int  tid    = threadIdx.x;
  const int  lane   = tid & 31;
  const int  wv     = tid >> 5;
  const int  ciw    = lane & 15;               // col-in-tile / row-in-tile
  const bool hiHalf = (lane >= 16);

  // ---- Stage 1: gather neighbor + node embeddings into LDS -----------------
  for (int c = tid; c < 1024; c += 256) {      // 32 rows x 32 float4 chunks
    const int row = c >> 5;
    const int q   = c & 31;
    const int v   = nidx[n * 32 + row];
    const float4 val = ((const float4*)(emb + (long long)v * 128))[q];
    ((float4*)&s_neigh[row][0])[q] = val;
    const int d0 = q * 4;
    s_xbf[row][d0 + 0] = f32_to_bf16_rne(val.x);
    s_xbf[row][d0 + 1] = f32_to_bf16_rne(val.y);
    s_xbf[row][d0 + 2] = f32_to_bf16_rne(val.z);
    s_xbf[row][d0 + 3] = f32_to_bf16_rne(val.w);
  }
  if (tid < 32) {
    const int v = nodes[n];
    const float4 val = ((const float4*)(emb + (long long)v * 128))[tid];
    const int d0 = tid * 4;
    s_nodebf[d0 + 0] = f32_to_bf16_rne(val.x);
    s_nodebf[d0 + 1] = f32_to_bf16_rne(val.y);
    s_nodebf[d0 + 2] = f32_to_bf16_rne(val.z);
    s_nodebf[d0 + 3] = f32_to_bf16_rne(val.w);
  }
  __syncthreads();

  const int   col   = wv * 16 + ciw;
  const float bias1 = b1[col];
  const float bias2 = b2[col];

  // ---- Layer 1: relu([32x256] x [256x128] + b1) ---------------------------
  // A 16x32 bf16 fragment (lane): row = lane&15 (+16 for M-tile 1);
// kbase = kc*32 + (lane<16 ? 0 : 16); elems 0-15 -> K=kbase..+15.
// K>=128 comes from the node embedding,
  // which is row-invariant -> broadcast read of s_nodebf.
  v8f acc0 = {};
  v8f acc1 = {};
#pragma unroll
  for (int kc = 0; kc < 8; ++kc) {
    const unsigned short* bp = w1f + (((kc * 8 + wv) * 32 + lane) << 4);
    const v16bf B = cat8(*(const v8bf*)bp, *(const v8bf*)(bp + 8));
    const int kbase = kc * 32 + (hiHalf ? 16 : 0);
    const unsigned short* a0 =
        (kc < 4) ? &s_xbf[ciw][kbase] : &s_nodebf[kbase - 128];
    const unsigned short* a1 = (kc < 4) ? &s_xbf[16 + ciw][kbase] : a0;
    const v16bf A0 = cat8(*(const v8bf*)a0, *(const v8bf*)(a0 + 8));
    const v16bf A1 = cat8(*(const v8bf*)a1, *(const v8bf*)(a1 + 8));
    acc0 = __builtin_amdgcn_wmma_f32_16x16x32_bf16(false, A0, false, B,
                                                   (short)0, acc0, false, false);
    acc1 = __builtin_amdgcn_wmma_f32_16x16x32_bf16(false, A1, false, B,
                                                   (short)0, acc1, false, false);
  }
  // D layout: VGPR v -> M = v + 8*(lane>=16); N = col.
#pragma unroll
  for (int v = 0; v < 8; ++v) {
    const int r = v + (hiHalf ? 8 : 0);
    s_hbf[r][col]      = f32_to_bf16_rne(fmaxf(acc0[v] + bias1, 0.f));
    s_hbf[16 + r][col] = f32_to_bf16_rne(fmaxf(acc1[v] + bias1, 0.f));
  }
  __syncthreads();

  // ---- Layer 2: relu([32x128] x [128x128] + b2) ---------------------------
  v8f acc2 = {};
  v8f acc3 = {};
#pragma unroll
  for (int kc = 0; kc < 4; ++kc) {
    const unsigned short* bp = w2f + (((kc * 8 + wv) * 32 + lane) << 4);
    const v16bf B = cat8(*(const v8bf*)bp, *(const v8bf*)(bp + 8));
    const int kbase = kc * 32 + (hiHalf ? 16 : 0);
    const unsigned short* a0 = &s_hbf[ciw][kbase];
    const unsigned short* a1 = &s_hbf[16 + ciw][kbase];
    const v16bf A0 = cat8(*(const v8bf*)a0, *(const v8bf*)(a0 + 8));
    const v16bf A1 = cat8(*(const v8bf*)a1, *(const v8bf*)(a1 + 8));
    acc2 = __builtin_amdgcn_wmma_f32_16x16x32_bf16(false, A0, false, B,
                                                   (short)0, acc2, false, false);
    acc3 = __builtin_amdgcn_wmma_f32_16x16x32_bf16(false, A1, false, B,
                                                   (short)0, acc3, false, false);
  }
#pragma unroll
  for (int v = 0; v < 8; ++v) {
    const int r = v + (hiHalf ? 8 : 0);
    s_h2[r][col]      = fmaxf(acc2[v] + bias2, 0.f);
    s_h2[16 + r][col] = fmaxf(acc3[v] + bias2, 0.f);
  }
  __syncthreads();

  // ---- Scores (h2 . W3 + b3) and softmax over 32 neighbors (wave 0) -------
  if (tid < 32) {
    float s = b3[0];
#pragma unroll 4
    for (int d = 0; d < 128; ++d) s += s_h2[tid][d] * W3[d];
    float mx = s;
#pragma unroll
    for (int off = 16; off > 0; off >>= 1)
      mx = fmaxf(mx, __shfl_xor(mx, off, 32));
    const float e = __expf(s - mx);
    float sum = e;
#pragma unroll
    for (int off = 16; off > 0; off >>= 1) sum += __shfl_xor(sum, off, 32);
    s_att[tid] = e / sum;
  }
  __syncthreads();

  // ---- Weighted aggregation: out[n] = sum_k att[k] * neigh[k] -------------
  if (tid < 128) {
    float a = 0.f;
#pragma unroll 8
    for (int k = 0; k < 32; ++k) a += s_att[k] * s_neigh[k][tid];
    out[(long long)n * 128 + tid] = a;
  }
}

// ---------------------------------------------------------------------------
// Launch
// ---------------------------------------------------------------------------
extern "C" void kernel_launch(void* const* d_in, const int* in_sizes, int n_in,
                              void* d_out, int out_size, void* d_ws,
                              size_t ws_size, hipStream_t stream) {
  const float* emb   = (const float*)d_in[0];  // [200000,128]
  const int*   nodes = (const int*)d_in[1];    // [N]
  const int*   nidx  = (const int*)d_in[2];    // [N,32]
  const float* W1    = (const float*)d_in[3];  // [256,128]
  const float* b1    = (const float*)d_in[4];  // [128]
  const float* W2    = (const float*)d_in[5];  // [128,128]
  const float* b2    = (const float*)d_in[6];  // [128]
  const float* W3    = (const float*)d_in[7];  // [128,1]
  const float* b3    = (const float*)d_in[8];  // [1]

  unsigned short* w1f = (unsigned short*)d_ws;        // 64 KB
  unsigned short* w2f = w1f + 32768;                  // 32 KB
  (void)ws_size; (void)n_in; (void)out_size;

  prep_weights<<<192, 256, 0, stream>>>(W1, W2, w1f, w2f);

  const int n_nodes = in_sizes[1];                    // 20000
  social_agg_wmma<<<n_nodes, 256, 0, stream>>>(emb, nodes, nidx, w1f, b1, w2f,
                                               b2, W3, b3, (float*)d_out);
}
